// MultiHeadAttention_37452114821901
// MI455X (gfx1250) — matrix operand order repaired and checked by reading, not compiled, not run on hardware
//
#include <hip/hip_runtime.h>

typedef __attribute__((ext_vector_type(16))) _Float16 v16h;
typedef __attribute__((ext_vector_type(8)))  float    v8f;
typedef __attribute__((ext_vector_type(4)))  _Float16 v4h;
typedef __attribute__((ext_vector_type(4)))  unsigned int u32x4;

union FragH { v16h h; u32x4 q[2]; };

__device__ __forceinline__ v8f wmma_f32_f16(v16h a, v16h b, v8f c) {
  // D = A(16x32 f16) x B(32x16 f16) + C(16x16 f32)
  return __builtin_amdgcn_wmma_f32_16x16x32_f16(false, a, false, b, (short)0, c,
                                                false, false);
}

#define S_LEN  2048
#define DMODEL 768
#define NHEAD  8
#define DEPTH  96
#define BATCH  4
#define MROWS  8192 /* BATCH*S_LEN */

#define TM 64
#define TN 64
#define TK 32
#define LDA 40 /* padded f16 row stride of A tile (64x32) */
#define LDB 72 /* padded f16 row stride of B tile (32x64) */

// ---------------------------------------------------------------------------
// Fused QKV projection: P = X @ W + b  (f32 in, f16 head-split out).
// z = 0:Q (layout [b,h,s,d]), 1:K ([b,h,s,d]), 2:V (transposed [b,h,d,s]).
// ---------------------------------------------------------------------------
__global__ __launch_bounds__(256) void proj_qkv_kernel(
    const float* __restrict__ q_in, const float* __restrict__ k_in,
    const float* __restrict__ v_in, const float* __restrict__ wq,
    const float* __restrict__ wk, const float* __restrict__ wv,
    const float* __restrict__ bq, const float* __restrict__ bk,
    const float* __restrict__ bv, _Float16* __restrict__ Qh,
    _Float16* __restrict__ Kt, _Float16* __restrict__ Vh) {
  __shared__ _Float16 lA[TM * LDA];
  __shared__ _Float16 lB[TK * LDB];

  const int z = blockIdx.z;
  const float* X  = (z == 0) ? q_in : ((z == 1) ? k_in : v_in);
  const float* W  = (z == 0) ? wq : ((z == 1) ? wk : wv);
  const float* Bp = (z == 0) ? bq : ((z == 1) ? bk : bv);

  const int m0 = blockIdx.y * TM;
  const int n0 = blockIdx.x * TN;
  const int tid = threadIdx.x;
  const int lane = tid & 31;
  const int wid = tid >> 5;     // 8 waves: 4 (m) x 2 (n)
  const int wr = wid & 3;
  const int wc = wid >> 2;
  const int hf = lane >> 4;
  const int ln = lane & 15;

  v8f c0 = {};
  v8f c1 = {};
  for (int k0 = 0; k0 < DMODEL; k0 += TK) {
    __syncthreads();
    {  // stage A tile 64x32 (f32 -> f16), 32 rows/pass, 8 threads/row
      const int r = tid >> 3;
      const int cc = (tid & 7) * 4;
#pragma unroll
      for (int p = 0; p < 2; ++p) {
        const float4 xv =
            *(const float4*)(X + (size_t)(m0 + r + p * 32) * DMODEL + k0 + cc);
        v4h hv;
        hv[0] = (_Float16)xv.x; hv[1] = (_Float16)xv.y;
        hv[2] = (_Float16)xv.z; hv[3] = (_Float16)xv.w;
        *(v4h*)(lA + (r + p * 32) * LDA + cc) = hv;
      }
    }
    {  // stage B tile 32x64 (f32 -> f16), 16 rows/pass, 16 threads/row
      const int r = tid >> 4;
      const int cc = (tid & 15) * 4;
#pragma unroll
      for (int p = 0; p < 2; ++p) {
        const float4 wv4 =
            *(const float4*)(W + (size_t)(k0 + r + p * 16) * DMODEL + n0 + cc);
        v4h hv;
        hv[0] = (_Float16)wv4.x; hv[1] = (_Float16)wv4.y;
        hv[2] = (_Float16)wv4.z; hv[3] = (_Float16)wv4.w;
        *(v4h*)(lB + (r + p * 16) * LDB + cc) = hv;
      }
    }
    __syncthreads();
    // A fragment: lane = M row, K striped as two 16B chunks per half-wave
    FragH a, b0, b1;
    const _Float16* ap = lA + (wr * 16 + ln) * LDA + hf * 8;
    a.q[0] = *(const u32x4*)ap;
    a.q[1] = *(const u32x4*)(ap + 16);
    // B fragments: lane = N col, K in table order per half-wave
    const _Float16* bp = lB + wc * 32 + ln;
    for (int s = 0; s < 16; ++s) {
      const int kk = (s < 8) ? (hf * 8 + s) : (16 + hf * 8 + s - 8);
      b0.h[s] = bp[kk * LDB];
      b1.h[s] = bp[kk * LDB + 16];
    }
    c0 = wmma_f32_f16(a.h, b0.h, c0);
    c1 = wmma_f32_f16(a.h, b1.h, c1);
  }

  // C/D layout: lane = N col (0..15), VGPR v -> M = v + 8*half
  const int mbase = m0 + wr * 16 + 8 * hf;
#pragma unroll
  for (int t = 0; t < 2; ++t) {
    const int n = n0 + wc * 32 + t * 16 + ln;
    const float bias = Bp[n];
    const int h = n / DEPTH;
    const int d = n % DEPTH;
    const v8f cc = t ? c1 : c0;
#pragma unroll
    for (int v = 0; v < 8; ++v) {
      const int m = mbase + v;
      const int b = m >> 11;
      const int s = m & (S_LEN - 1);
      const float val = cc[v] + bias;
      if (z == 1) {
        Kt[(((size_t)b * NHEAD + h) * S_LEN + s) * DEPTH + d] = (_Float16)val;
      } else if (z == 0) {
        Qh[(((size_t)b * NHEAD + h) * S_LEN + s) * DEPTH + d] = (_Float16)val;
      } else {
        Vh[(((size_t)b * NHEAD + h) * DEPTH + d) * S_LEN + s] = (_Float16)val;
      }
    }
  }
}

// ---------------------------------------------------------------------------
// Fused attention: per (b,h, 16-row block): logits -> softmax -> attn(write)
// -> ctx = attn @ V.   256 threads = 8 wave32.
// ---------------------------------------------------------------------------
__global__ __launch_bounds__(256) void attention_kernel(
    const _Float16* __restrict__ Qh, const _Float16* __restrict__ Kt,
    const _Float16* __restrict__ Vh, const float* __restrict__ mask,
    float* __restrict__ attn_out, _Float16* __restrict__ Ctx) {
  extern __shared__ char smem[];
  float* attnF = (float*)smem;                        // [16][2048] f32 128KB
  _Float16* attnH = (_Float16*)(smem + 131072);       // [16][2048] f16  64KB
  float* red = (float*)(smem + 131072 + 65536);       // [16][16]
  float* rowmax = red + 256;                          // [16]
  float* rowinv = rowmax + 16;                        // [16]

  const int m0 = blockIdx.x * 16;
  const int bh = blockIdx.y;
  const int b = bh >> 3;
  const int tid = threadIdx.x;
  const int lane = tid & 31;
  const int wid = tid >> 5;
  const int hf = lane >> 4;
  const int ln = lane & 15;
  const float SCALE = 0.10206207261596575f;  // 1/sqrt(96)

  // ---- phase 1: logits = Q Kt (per wave: 16 column tiles of 16) ----
  FragH aq[3];
#pragma unroll
  for (int kc = 0; kc < 3; ++kc) {
    const _Float16* ap =
        Qh + ((size_t)bh * S_LEN + m0 + ln) * DEPTH + kc * 32 + hf * 8;
    aq[kc].q[0] = *(const u32x4*)ap;
    aq[kc].q[1] = *(const u32x4*)(ap + 16);
  }
  for (int jj = 0; jj < 16; ++jj) {
    const int j = wid * 16 + jj;
    const int col = j * 16 + ln;
    v8f c = {};
#pragma unroll
    for (int kc = 0; kc < 3; ++kc) {
      FragH bkf;  // B frag: lane = key col, K in table order
      const _Float16* bp =
          Kt + ((size_t)bh * S_LEN + j * 16 + ln) * DEPTH + kc * 32 + hf * 8;
      bkf.q[0] = *(const u32x4*)bp;
      bkf.q[1] = *(const u32x4*)(bp + 16);
      c = wmma_f32_f16(aq[kc].h, bkf.h, c);
    }
    const float mval = mask[(size_t)b * S_LEN + col] * -1.0e9f;
#pragma unroll
    for (int v = 0; v < 8; ++v) {
      attnF[(v + 8 * hf) * S_LEN + col] = c[v] * SCALE + mval;
    }
  }
  __syncthreads();

  // ---- phase 2: row softmax (16 rows x 2048, 16 threads per row) ----
  {
    const int r = tid >> 4;
    const int seg = tid & 15;
    float* rowp = attnF + r * S_LEN + seg * 128;
    float mx = -3.0e38f;
    for (int i = 0; i < 128; ++i) mx = fmaxf(mx, rowp[i]);
    red[r * 16 + seg] = mx;
    __syncthreads();
    if (seg == 0) {
      float m2 = red[r * 16];
#pragma unroll
      for (int i = 1; i < 16; ++i) m2 = fmaxf(m2, red[r * 16 + i]);
      rowmax[r] = m2;
    }
    __syncthreads();
    const float rm = rowmax[r];
    float sum = 0.0f;
    for (int i = 0; i < 128; ++i) {
      const float e = __expf(rowp[i] - rm);
      rowp[i] = e;
      sum += e;
    }
    red[r * 16 + seg] = sum;
    __syncthreads();
    if (seg == 0) {
      float s2 = 0.0f;
#pragma unroll
      for (int i = 0; i < 16; ++i) s2 += red[r * 16 + i];
      rowinv[r] = 1.0f / s2;
    }
    __syncthreads();
  }
  // phase 2b: normalize, coalesced f32 attn store, f16 copy for attn@V
  {
    float* gbase = attn_out + ((size_t)bh * S_LEN + m0) * S_LEN;
    for (int idx = tid; idx < 16 * 512; idx += 256) {
      const int row = idx >> 9;
      const int c4 = idx & 511;
      float4 vv = ((const float4*)(attnF + row * S_LEN))[c4];
      const float inv = rowinv[row];
      vv.x *= inv; vv.y *= inv; vv.z *= inv; vv.w *= inv;
      ((float4*)(gbase + (size_t)row * S_LEN))[c4] = vv;
      v4h hv;
      hv[0] = (_Float16)vv.x; hv[1] = (_Float16)vv.y;
      hv[2] = (_Float16)vv.z; hv[3] = (_Float16)vv.w;
      *(v4h*)(attnH + row * S_LEN + c4 * 4) = hv;
    }
  }
  __syncthreads();

  // ---- phase 3: ctx[16x96] = attn[16x2048] @ V[2048x96] (6 waves) ----
  if (wid < 6) {
    const int n0 = wid * 16;
    v8f c = {};
    for (int k0 = 0; k0 < S_LEN; k0 += 32) {
      FragH a, bvf;
      const _Float16* ap = attnH + ln * S_LEN + k0 + hf * 8;
      a.q[0] = *(const u32x4*)ap;
      a.q[1] = *(const u32x4*)(ap + 16);
      const _Float16* bp = Vh + ((size_t)bh * DEPTH + n0 + ln) * S_LEN + k0 + hf * 8;
      bvf.q[0] = *(const u32x4*)bp;
      bvf.q[1] = *(const u32x4*)(bp + 16);
      c = wmma_f32_f16(a.h, bvf.h, c);
    }
    const int h = bh & 7;
    const int n = n0 + ln;
#pragma unroll
    for (int v = 0; v < 8; ++v) {
      const int s = m0 + v + 8 * hf;
      Ctx[(((size_t)b * S_LEN + s) * NHEAD + h) * DEPTH + n] = (_Float16)c[v];
    }
  }
}

// ---------------------------------------------------------------------------
// Output projection: out = ctx @ wo + bo  (f16 A, f32 W/out).
// ---------------------------------------------------------------------------
__global__ __launch_bounds__(256) void proj_out_kernel(
    const _Float16* __restrict__ Ctx, const float* __restrict__ wo,
    const float* __restrict__ bo, float* __restrict__ out) {
  __shared__ _Float16 lA[TM * LDA];
  __shared__ _Float16 lB[TK * LDB];

  const int m0 = blockIdx.y * TM;
  const int n0 = blockIdx.x * TN;
  const int tid = threadIdx.x;
  const int lane = tid & 31;
  const int wid = tid >> 5;
  const int wr = wid & 3;
  const int wc = wid >> 2;
  const int hf = lane >> 4;
  const int ln = lane & 15;

  v8f c0 = {};
  v8f c1 = {};
  for (int k0 = 0; k0 < DMODEL; k0 += TK) {
    __syncthreads();
    {  // stage A tile 64x32 f16 directly, one pass (16B per thread)
      const int r = tid >> 2;
      const int cc = (tid & 3) * 8;
      const u32x4 xv =
          *(const u32x4*)(Ctx + (size_t)(m0 + r) * DMODEL + k0 + cc);
      *(u32x4*)(lA + r * LDA + cc) = xv;
    }
    {  // stage B tile 32x64 (f32 -> f16)
      const int r = tid >> 4;
      const int cc = (tid & 15) * 4;
#pragma unroll
      for (int p = 0; p < 2; ++p) {
        const float4 wv4 =
            *(const float4*)(wo + (size_t)(k0 + r + p * 16) * DMODEL + n0 + cc);
        v4h hv;
        hv[0] = (_Float16)wv4.x; hv[1] = (_Float16)wv4.y;
        hv[2] = (_Float16)wv4.z; hv[3] = (_Float16)wv4.w;
        *(v4h*)(lB + (r + p * 16) * LDB + cc) = hv;
      }
    }
    __syncthreads();
    FragH a, b0, b1;
    const _Float16* ap = lA + (wr * 16 + ln) * LDA + hf * 8;
    a.q[0] = *(const u32x4*)ap;
    a.q[1] = *(const u32x4*)(ap + 16);
    const _Float16* bp = lB + wc * 32 + ln;
    for (int s = 0; s < 16; ++s) {
      const int kk = (s < 8) ? (hf * 8 + s) : (16 + hf * 8 + s - 8);
      b0.h[s] = bp[kk * LDB];
      b1.h[s] = bp[kk * LDB + 16];
    }
    c0 = wmma_f32_f16(a.h, b0.h, c0);
    c1 = wmma_f32_f16(a.h, b1.h, c1);
  }

  const int mbase = m0 + wr * 16 + 8 * hf;
#pragma unroll
  for (int t = 0; t < 2; ++t) {
    const int n = n0 + wc * 32 + t * 16 + ln;
    const float bias = bo[n];
    const v8f cc = t ? c1 : c0;
#pragma unroll
    for (int v = 0; v < 8; ++v) {
      out[(size_t)(mbase + v) * DMODEL + n] = cc[v] + bias;
    }
  }
}

extern "C" void kernel_launch(void* const* d_in, const int* in_sizes, int n_in,
                              void* d_out, int out_size, void* d_ws,
                              size_t ws_size, hipStream_t stream) {
  (void)in_sizes; (void)n_in; (void)out_size; (void)ws_size;
  const float* q_in = (const float*)d_in[0];
  const float* k_in = (const float*)d_in[1];
  const float* v_in = (const float*)d_in[2];
  const float* mask = (const float*)d_in[3];
  const float* wq = (const float*)d_in[4];
  const float* bq = (const float*)d_in[5];
  const float* wk = (const float*)d_in[6];
  const float* bk = (const float*)d_in[7];
  const float* wv = (const float*)d_in[8];
  const float* bv = (const float*)d_in[9];
  const float* wo = (const float*)d_in[10];
  const float* bo = (const float*)d_in[11];

  float* out = (float*)d_out;
  float* attn_out = out + (size_t)MROWS * DMODEL;  // second tuple element

  const size_t hsz = (size_t)BATCH * NHEAD * S_LEN * DEPTH;  // 6.29M f16
  _Float16* Qh = (_Float16*)d_ws;
  _Float16* Kt = Qh + hsz;
  _Float16* Vh = Kt + hsz;
  _Float16* Ctx = Vh + hsz;

  proj_qkv_kernel<<<dim3(DMODEL / TN, MROWS / TM, 3), 256, 0, stream>>>(
      q_in, k_in, v_in, wq, wk, wv, bq, bk, bv, Qh, Kt, Vh);

  const int attn_lds = 16 * 2048 * 4 + 16 * 2048 * 2 + 16 * 16 * 4 + 64 + 64;
  attention_kernel<<<dim3(S_LEN / 16, BATCH * NHEAD), 256, attn_lds, stream>>>(
      Qh, Kt, Vh, mask, attn_out, Ctx);

  proj_out_kernel<<<dim3(DMODEL / TN, MROWS / TM), 256, 0, stream>>>(Ctx, wo,
                                                                     bo, out);
}
